// HNet_61297773248716
// MI455X (gfx1250) — matrix operand order repaired and checked by reading, not compiled, not run on hardware
//
#include <hip/hip_runtime.h>
#include <cstddef>

#define BATCH   8
#define SEQL    1024
#define VOCABSZ 260
#define DENC    128
#define DMAIN   256
#define NHEADSN 4
#define ROWS    (BATCH * SEQL)

typedef __attribute__((ext_vector_type(16))) _Float16 v16h;
typedef __attribute__((ext_vector_type(8)))  _Float16 v8h;
typedef __attribute__((ext_vector_type(8)))  float    v8f;

// ---------------------------------------------------------------------------
// WMMA helpers (gfx1250: V_WMMA_F32_16X16X32_F16, wave32)
// ---------------------------------------------------------------------------
__device__ __forceinline__ v8f wmma_f32_f16(v16h a, v16h b, v8f c) {
  return __builtin_amdgcn_wmma_f32_16x16x32_f16(false, a, false, b, (short)0, c,
                                                false, false);
}

// A-fragment (16x32 f16) from packed-f16 memory (global or LDS): two 16B segs.
// Layout, lane half g: elements 0..15 -> K = g*16+0..15 (contiguous)
__device__ __forceinline__ v16h load_afrag_h(const _Float16* p, int g) {
  v8h s0 = *reinterpret_cast<const v8h*>(p + g * 16);
  v8h s1 = *reinterpret_cast<const v8h*>(p + g * 16 + 8);
  v16h f;
  f[0] = s0[0]; f[1] = s0[1]; f[2]  = s0[2]; f[3]  = s0[3];
  f[4] = s0[4]; f[5] = s0[5]; f[6]  = s0[6]; f[7]  = s0[7];
  f[8] = s1[0]; f[9] = s1[1]; f[10] = s1[2]; f[11] = s1[3];
  f[12] = s1[4]; f[13] = s1[5]; f[14] = s1[6]; f[15] = s1[7];
  return f;
}

__device__ __forceinline__ float gelu_exact(float x) {
  return 0.5f * x * (1.0f + erff(x * 0.70710678118654752f));
}
__device__ __forceinline__ float wave_red_sum16(float v) {
  v += __shfl_xor(v, 1, 32); v += __shfl_xor(v, 2, 32);
  v += __shfl_xor(v, 4, 32); v += __shfl_xor(v, 8, 32);
  return v;
}
__device__ __forceinline__ float wave_red_max16(float v) {
  v = fmaxf(v, __shfl_xor(v, 1, 32)); v = fmaxf(v, __shfl_xor(v, 2, 32));
  v = fmaxf(v, __shfl_xor(v, 4, 32)); v = fmaxf(v, __shfl_xor(v, 8, 32));
  return v;
}
__device__ __forceinline__ float wave_red_sum32(float v) {
  v += __shfl_xor(v, 1, 32);  v += __shfl_xor(v, 2, 32);
  v += __shfl_xor(v, 4, 32);  v += __shfl_xor(v, 8, 32);
  v += __shfl_xor(v, 16, 32);
  return v;
}

// ---------------------------------------------------------------------------
// WMMA GEMM: C[M,N] = A[M,K] @ W[N,K]^T.  A and W packed f16.
// Block 256 thr (8 waves); tile 128(M) x 64(N); K staged in 128-chunks.
// Per wave/ks: 1 A-frag (2 global b128) -> 4 B-frags (8 ds b128, clustered)
// -> 4 back-to-back WMMAs.
// FUSE: 0 none, 1 exact GELU, 2 += R(f32).  OUTH: 1 -> f16 out, 0 -> f32 out.
// ---------------------------------------------------------------------------
template<int FUSE, int OUTH>
__global__ void __launch_bounds__(256) HNet_gemm_kernel(
    const _Float16* __restrict__ A, const _Float16* __restrict__ W,
    const float* R, float* Cf, _Float16* Ch, int M, int N, int K)
{
  __shared__ __attribute__((aligned(32))) _Float16 ldsW[64 * 128];  // 16 KB
  const int tid = threadIdx.x;
  const int n0 = blockIdx.x * 64;
  const int wave = tid >> 5, lane = tid & 31, g = lane >> 4, hl = lane & 15;
  const int m0 = blockIdx.y * 128 + wave * 16;
  const _Float16* arow = A + (size_t)(m0 + hl) * K;
  const v8f vz = {};
  v8f acc[4];
  acc[0] = vz; acc[1] = vz; acc[2] = vz; acc[3] = vz;

  for (int kc = 0; kc < K; kc += 128) {
    if (kc) __syncthreads();
    for (int v = tid; v < 64 * 16; v += 256) {
      int n = v >> 4, kg = v & 15;
      float4 d;
      if (n0 + n < N)
        d = *reinterpret_cast<const float4*>(&W[(size_t)(n0 + n) * K + kc + kg * 8]);
      else
        d = make_float4(0.f, 0.f, 0.f, 0.f);
      *reinterpret_cast<float4*>(&ldsW[n * 128 + kg * 8]) = d;
    }
    __syncthreads();
#pragma unroll
    for (int ks = 0; ks < 4; ++ks) {
      v16h af = load_afrag_h(arow + kc + ks * 32, g);
      const _Float16* bbase = &ldsW[hl * 128 + ks * 32 + g * 16];
      v16h bf0 = *reinterpret_cast<const v16h*>(bbase);
      v16h bf1 = *reinterpret_cast<const v16h*>(bbase + 16 * 128);
      v16h bf2 = *reinterpret_cast<const v16h*>(bbase + 32 * 128);
      v16h bf3 = *reinterpret_cast<const v16h*>(bbase + 48 * 128);
      acc[0] = wmma_f32_f16(af, bf0, acc[0]);
      acc[1] = wmma_f32_f16(af, bf1, acc[1]);
      acc[2] = wmma_f32_f16(af, bf2, acc[2]);
      acc[3] = wmma_f32_f16(af, bf3, acc[3]);
    }
  }
#pragma unroll
  for (int nt = 0; nt < 4; ++nt) {
    int col = n0 + nt * 16 + hl;
    if (col < N) {
#pragma unroll
      for (int r = 0; r < 8; ++r) {
        size_t row = (size_t)m0 + g * 8 + r;
        float v = acc[nt][r];
        if (FUSE == 1) v = gelu_exact(v);
        if (FUSE == 2) v += R[row * N + col];
        if (OUTH) Ch[row * N + col] = (_Float16)v;
        else      Cf[row * N + col] = v;
      }
    }
  }
}

// ---------------------------------------------------------------------------
// Flash attention, one block per (b, h, 128 q rows); 8 waves x 16 q rows.
// qkv: packed f16 (B, S, 3*D), q|k|v concat. Output f16. DH = 32 or 64.
// ---------------------------------------------------------------------------
template<int DH>
__global__ void __launch_bounds__(256) HNet_attn_kernel(
    const _Float16* __restrict__ qkv, const float* __restrict__ maskf,
    _Float16* __restrict__ out, int useMask)
{
  constexpr int D = NHEADSN * DH;
  constexpr int D3 = 3 * D;
  constexpr int DSTEPS = DH / 32;
  constexpr int DTILES = DH / 16;
  constexpr int KVEC = (32 * DH) / 8;  // vec8 groups per 32xDH tile
  const int S = SEQL;
  const int tid = threadIdx.x;
  const int wave = tid >> 5, lane = tid & 31, g = lane >> 4, hl = lane & 15;
  const int b = blockIdx.y / NHEADSN, h = blockIdx.y % NHEADSN;
  const int q0 = blockIdx.x * 128;
  const int q0w = q0 + wave * 16;
  const _Float16* qbase = qkv + (size_t)b * S * D3 + h * DH;
  const _Float16* kbase = qbase + D;
  const _Float16* vbase = qbase + 2 * D;

  __shared__ __attribute__((aligned(32))) _Float16 ldsK[32 * DH];   // [key][d]
  __shared__ __attribute__((aligned(32))) _Float16 ldsVt[DH * 32];  // [d][key]
  __shared__ __attribute__((aligned(32))) _Float16 ldsP[8][16 * 32];

  const v8f vzero = {};
  const float scale = rsqrtf((float)DH);
  v16h aq[DSTEPS];
  {
    const _Float16* qr = qbase + (size_t)(q0w + hl) * D3;
#pragma unroll
    for (int st = 0; st < DSTEPS; ++st) aq[st] = load_afrag_h(qr + st * 32, g);
  }
  float mrow[8], lsum[8];
  v8f oacc[DTILES];
#pragma unroll
  for (int r = 0; r < 8; ++r) { mrow[r] = -3.0e38f; lsum[r] = 0.f; }
#pragma unroll
  for (int dt = 0; dt < DTILES; ++dt) oacc[dt] = vzero;

  const int nkt = (q0 + 128) / 32;
  for (int ki = 0; ki < nkt; ++ki) {
    const int key0 = ki * 32;
    // stage K (row-major copy, 16B vectors) and V (transposed) as f16
    for (int v = tid; v < KVEC; v += 256) {
      int kr = v / (DH / 8), dg = v - kr * (DH / 8);
      size_t src = (size_t)(key0 + kr) * D3 + dg * 8;
      *reinterpret_cast<v8h*>(&ldsK[kr * DH + dg * 8]) =
          *reinterpret_cast<const v8h*>(&kbase[src]);
      v8h vv = *reinterpret_cast<const v8h*>(&vbase[src]);
#pragma unroll
      for (int e = 0; e < 8; ++e) ldsVt[(dg * 8 + e) * 32 + kr] = vv[e];
    }
    __syncthreads();
    if (key0 <= q0w + 15) {  // wave-uniform causal skip (EXEC stays all-1s)
      float pv[2][8];
#pragma unroll
      for (int nt = 0; nt < 2; ++nt) {
        v8f sc = vzero;
#pragma unroll
        for (int st = 0; st < DSTEPS; ++st) {
          v16h bk = *reinterpret_cast<const v16h*>(
              &ldsK[(nt * 16 + hl) * DH + st * 32 + g * 16]);
          sc = wmma_f32_f16(aq[st], bk, sc);
        }
        const int key = key0 + nt * 16 + hl;
        float mb = 0.f;
        if (useMask) mb = (maskf[(size_t)b * S + key] > 0.5f) ? 0.f : -1.0e9f;
#pragma unroll
        for (int r = 0; r < 8; ++r) {
          int qr = q0w + g * 8 + r;
          pv[nt][r] = sc[r] * scale + mb + ((key > qr) ? -1.0e9f : 0.f);
        }
      }
#pragma unroll
      for (int r = 0; r < 8; ++r) {
        float mloc = wave_red_max16(fmaxf(pv[0][r], pv[1][r]));
        float mnew = fmaxf(mrow[r], mloc);
        float alpha = __expf(mrow[r] - mnew);
        mrow[r] = mnew;
        float p0 = __expf(pv[0][r] - mnew);
        float p1 = __expf(pv[1][r] - mnew);
        float rs = wave_red_sum16(p0 + p1);
        lsum[r] = lsum[r] * alpha + rs;
#pragma unroll
        for (int dt = 0; dt < DTILES; ++dt) oacc[dt][r] *= alpha;
        ldsP[wave][(g * 8 + r) * 32 + hl] = (_Float16)p0;
        ldsP[wave][(g * 8 + r) * 32 + 16 + hl] = (_Float16)p1;
      }
      // P tile re-read is exactly an A-fragment: two ds_load_b128
      v16h ap = load_afrag_h(&ldsP[wave][hl * 32], g);
      v16h bv[DTILES];
#pragma unroll
      for (int dt = 0; dt < DTILES; ++dt)
        bv[dt] = *reinterpret_cast<const v16h*>(&ldsVt[(dt * 16 + hl) * 32 + g * 16]);
#pragma unroll
      for (int dt = 0; dt < DTILES; ++dt)
        oacc[dt] = wmma_f32_f16(ap, bv[dt], oacc[dt]);
    }
    __syncthreads();
  }
#pragma unroll
  for (int r = 0; r < 8; ++r) {
    float inv = 1.0f / lsum[r];
    size_t row = (size_t)b * S + q0w + g * 8 + r;
#pragma unroll
    for (int dt = 0; dt < DTILES; ++dt)
      out[row * D + h * DH + dt * 16 + hl] = (_Float16)(oacc[dt][r] * inv);
  }
}

// ---------------------------------------------------------------------------
// LayerNorm: one wave per row; f32 in, packed f16 out (feeds WMMA A operand).
// ---------------------------------------------------------------------------
template<int D>
__global__ void __launch_bounds__(256) HNet_ln_kernel(
    const float* __restrict__ X, const float* __restrict__ w,
    const float* __restrict__ bb, _Float16* __restrict__ Y)
{
  constexpr int NPL = D / 32;
  const int wave = threadIdx.x >> 5, lane = threadIdx.x & 31;
  const size_t row = (size_t)blockIdx.x * 8 + wave;
  const float* xr = X + row * D;
  float v[NPL], s = 0.f;
#pragma unroll
  for (int i = 0; i < NPL; ++i) { v[i] = xr[i * 32 + lane]; s += v[i]; }
  float mu = wave_red_sum32(s) * (1.0f / D);
  float var = 0.f;
#pragma unroll
  for (int i = 0; i < NPL; ++i) { float d = v[i] - mu; var += d * d; }
  var = wave_red_sum32(var) * (1.0f / D);
  float rs = rsqrtf(var + 1e-5f);
  _Float16* yr = Y + row * D;
#pragma unroll
  for (int i = 0; i < NPL; ++i) {
    int c = i * 32 + lane;
    yr[c] = (_Float16)((v[i] - mu) * rs * w[c] + bb[c]);
  }
}

// ---------------------------------------------------------------------------
// Small pipeline kernels
// ---------------------------------------------------------------------------
__global__ void HNet_cvt_kernel(const float* __restrict__ src,
                                _Float16* __restrict__ dst, int n) {
  for (int i = blockIdx.x * blockDim.x + threadIdx.x; i < n;
       i += gridDim.x * blockDim.x)
    dst[i] = (_Float16)src[i];
}

__global__ void HNet_embed_kernel(const int* __restrict__ ids,
                                  const float* __restrict__ emb,
                                  float* __restrict__ X) {
  int i = blockIdx.x, t = threadIdx.x;
  X[(size_t)i * DENC + t] = emb[(size_t)ids[i] * DENC + t];
}

__global__ void __launch_bounds__(256) HNet_route_kernel(
    const float* __restrict__ XQ, const float* __restrict__ XK,
    float* __restrict__ P, float* __restrict__ BH)
{
  const int wave = threadIdx.x >> 5, lane = threadIdx.x & 31;
  const int idx = blockIdx.x * 8 + wave;
  const int s = idx & (SEQL - 1);
  const float* qr = XQ + (size_t)idx * DENC;
  const float* kr = XK + (size_t)((s > 0) ? idx - 1 : idx) * DENC;
  float qq = 0.f, kk2 = 0.f, qk = 0.f;
#pragma unroll
  for (int i = 0; i < 4; ++i) {
    float q = qr[i * 32 + lane];
    float k = kr[i * 32 + lane];
    qq += q * q; kk2 += k * k; qk += q * k;
  }
  qq = wave_red_sum32(qq); kk2 = wave_red_sum32(kk2); qk = wave_red_sum32(qk);
  if (lane == 0) {
    float p;
    if (s == 0) p = 1.f;
    else {
      float c = qk / (fmaxf(sqrtf(qq), 1e-12f) * fmaxf(sqrtf(kk2), 1e-12f));
      p = fminf(fmaxf(0.5f * (1.f - c), 0.f), 1.f);
    }
    P[idx] = p;
    BH[idx] = (p >= 0.5f) ? 1.f : 0.f;
  }
}

__global__ void __launch_bounds__(256) HNet_scan_kernel(
    const float* __restrict__ BH, int* __restrict__ CIDX, float* __restrict__ MASKF)
{
  __shared__ int ps[256];
  const int b = blockIdx.x, t = threadIdx.x;
  const int base = b * SEQL + t * 4;
  int v0 = BH[base + 0] > 0.5f, v1 = BH[base + 1] > 0.5f;
  int v2 = BH[base + 2] > 0.5f, v3 = BH[base + 3] > 0.5f;
  int seg = v0 + v1 + v2 + v3;
  ps[t] = seg;
  __syncthreads();
  for (int off = 1; off < 256; off <<= 1) {
    int add = (t >= off) ? ps[t - off] : 0;
    __syncthreads();
    ps[t] += add;
    __syncthreads();
  }
  int run = ps[t] - seg;
  run += v0; CIDX[base + 0] = run - 1;
  run += v1; CIDX[base + 1] = run - 1;
  run += v2; CIDX[base + 2] = run - 1;
  run += v3; CIDX[base + 3] = run - 1;
  int count = ps[255];
#pragma unroll
  for (int j = 0; j < 4; ++j)
    MASKF[base + j] = ((t * 4 + j) < count) ? 1.f : 0.f;
}

__global__ void HNet_scatter_kernel(const float* __restrict__ X,
                                    const float* __restrict__ P,
                                    const float* __restrict__ BH,
                                    const int* __restrict__ CIDX,
                                    _Float16* __restrict__ CMPh,
                                    float* __restrict__ PC) {
  int i = blockIdx.x, t = threadIdx.x;
  if (BH[i] > 0.5f) {
    int b = i >> 10;
    int j = CIDX[i];
    CMPh[((size_t)(b * SEQL + j)) * DENC + t] = (_Float16)X[(size_t)i * DENC + t];
    if (t == 0) PC[b * SEQL + j] = P[i];
  }
}

__global__ void HNet_fill_kernel(float* __restrict__ p, int n, float v) {
  for (int i = blockIdx.x * blockDim.x + threadIdx.x; i < n;
       i += gridDim.x * blockDim.x)
    p[i] = v;
}
__global__ void HNet_fillh_kernel(_Float16* __restrict__ p, int n) {
  for (int i = blockIdx.x * blockDim.x + threadIdx.x; i < n;
       i += gridDim.x * blockDim.x)
    p[i] = (_Float16)0.f;
}

__global__ void __launch_bounds__(256) HNet_reduce_kernel(
    const float* __restrict__ BH, const float* __restrict__ P, float* __restrict__ RED) {
  float a = 0.f, g = 0.f;
  for (int i = blockIdx.x * blockDim.x + threadIdx.x; i < ROWS;
       i += gridDim.x * blockDim.x) { a += BH[i]; g += P[i]; }
  a = wave_red_sum32(a); g = wave_red_sum32(g);
  if ((threadIdx.x & 31) == 0) { atomicAdd(&RED[0], a); atomicAdd(&RED[1], g); }
}

__global__ void HNet_loss_kernel(const float* __restrict__ RED, float* __restrict__ outp) {
  float F = RED[0] * (1.0f / (float)ROWS);
  float G = RED[1] * (1.0f / (float)ROWS);
  const float Nr = 6.0f;
  outp[0] = Nr / (Nr - 1.f) * ((Nr - 1.f) * F * G + (1.f - F) * (1.f - G));
}

__global__ void HNet_ema_kernel(const float* __restrict__ ZD,
                                const float* __restrict__ PC, float* __restrict__ ZB) {
  const int b = blockIdx.x, f = threadIdx.x;
  float z = 0.f;
  for (int j = 0; j < SEQL; ++j) {
    size_t r = (size_t)(b * SEQL + j);
    float zd = ZD[r * DENC + f];
    float pc = fminf(fmaxf(PC[r], 1e-4f), 1.f - 1e-4f);
    float a = (j == 0) ? 0.f : (1.f - pc);
    float bv = (j == 0) ? zd : pc * zd;
    z = a * z + bv;
    ZB[r * DENC + f] = z;
  }
}

__global__ void HNet_addup_kernel(float* __restrict__ X, const float* __restrict__ ZB,
                                  const int* __restrict__ CIDX) {
  int i = blockIdx.x, t = threadIdx.x;
  int b = i >> 10;
  int j = CIDX[i];
  X[(size_t)i * DENC + t] += ZB[((size_t)(b * SEQL + j)) * DENC + t];
}

// ---------------------------------------------------------------------------
// Host orchestration
// ---------------------------------------------------------------------------
struct WsBufs {
  float *X, *Z, *XQ, *XK, *ZD, *ZB, *Pp, *BH, *PC, *MASKF, *RED;
  int* CIDX;
  _Float16 *T0h, *QKVh, *ATTh, *Hh, *Xh, *CMPh;
};

static void run_block(hipStream_t stream, float* x, int d,
                      const float* ln1w, const float* ln1b,
                      const _Float16* qkvW, const _Float16* woW,
                      const float* ln2w, const float* ln2b,
                      const _Float16* w1W, const _Float16* w2W,
                      const float* maskf, const WsBufs& ws) {
  dim3 blk(256);
  dim3 gLN(ROWS / 8);
  if (d == DENC) HNet_ln_kernel<DENC><<<gLN, blk, 0, stream>>>(x, ln1w, ln1b, ws.T0h);
  else           HNet_ln_kernel<DMAIN><<<gLN, blk, 0, stream>>>(x, ln1w, ln1b, ws.T0h);
  HNet_gemm_kernel<0, 1><<<dim3((3 * d + 63) / 64, ROWS / 128), blk, 0, stream>>>(
      ws.T0h, qkvW, nullptr, nullptr, ws.QKVh, ROWS, 3 * d, d);
  int useMask = (maskf != nullptr) ? 1 : 0;
  dim3 gAT(SEQL / 128, BATCH * NHEADSN);
  if (d == DENC) HNet_attn_kernel<32><<<gAT, blk, 0, stream>>>(ws.QKVh, maskf, ws.ATTh, useMask);
  else           HNet_attn_kernel<64><<<gAT, blk, 0, stream>>>(ws.QKVh, maskf, ws.ATTh, useMask);
  HNet_gemm_kernel<2, 0><<<dim3((d + 63) / 64, ROWS / 128), blk, 0, stream>>>(
      ws.ATTh, woW, x, x, nullptr, ROWS, d, d);
  if (d == DENC) HNet_ln_kernel<DENC><<<gLN, blk, 0, stream>>>(x, ln2w, ln2b, ws.T0h);
  else           HNet_ln_kernel<DMAIN><<<gLN, blk, 0, stream>>>(x, ln2w, ln2b, ws.T0h);
  HNet_gemm_kernel<1, 1><<<dim3((2 * d + 63) / 64, ROWS / 128), blk, 0, stream>>>(
      ws.T0h, w1W, nullptr, nullptr, ws.Hh, ROWS, 2 * d, d);
  HNet_gemm_kernel<2, 0><<<dim3((d + 63) / 64, ROWS / 128), blk, 0, stream>>>(
      ws.Hh, w2W, x, x, nullptr, ROWS, d, 2 * d);
}

extern "C" void kernel_launch(void* const* d_in, const int* in_sizes, int n_in,
                              void* d_out, int out_size, void* d_ws, size_t ws_size,
                              hipStream_t stream) {
  (void)in_sizes; (void)n_in; (void)out_size; (void)ws_size;
  const int*   byte_ids  = (const int*)d_in[0];
  const float* embed     = (const float*)d_in[1];
  const float* wq        = (const float*)d_in[2];
  const float* wk        = (const float*)d_in[3];
  const float* proj_up   = (const float*)d_in[4];
  const float* proj_down = (const float*)d_in[5];
  const float* norm_w    = (const float*)d_in[6];
  const float* norm_b    = (const float*)d_in[7];
  const float* enc[8];  for (int i = 0; i < 8; ++i) enc[i]  = (const float*)d_in[8 + i];
  const float* mn[8];   for (int i = 0; i < 8; ++i) mn[i]   = (const float*)d_in[16 + i];
  const float* dec[8];  for (int i = 0; i < 8; ++i) dec[i]  = (const float*)d_in[24 + i];

  WsBufs ws;
  float* fbase = (float*)d_ws;
  size_t fo = 0;
  auto falloc = [&](size_t n) { float* p = fbase + fo; fo += n; return p; };
  ws.X  = falloc((size_t)ROWS * DENC);
  ws.Z  = falloc((size_t)ROWS * DMAIN);
  ws.XQ = falloc((size_t)ROWS * DENC);
  ws.XK = falloc((size_t)ROWS * DENC);
  ws.ZD = falloc((size_t)ROWS * DENC);
  ws.ZB = falloc((size_t)ROWS * DENC);
  ws.Pp = falloc(ROWS); ws.BH = falloc(ROWS); ws.PC = falloc(ROWS);
  ws.MASKF = falloc(ROWS);
  ws.RED = falloc(8);
  ws.CIDX = (int*)falloc(ROWS);

  _Float16* hbase = (_Float16*)(fbase + ((fo + 7) & ~(size_t)7));  // 32B align
  size_t ho = 0;
  auto halloc = [&](size_t n) { _Float16* p = hbase + ho; ho += n; return p; };
  ws.T0h  = halloc((size_t)ROWS * DMAIN);
  ws.QKVh = halloc((size_t)ROWS * 3 * DMAIN);
  ws.ATTh = halloc((size_t)ROWS * DMAIN);
  ws.Hh   = halloc((size_t)ROWS * 2 * DMAIN);
  ws.Xh   = halloc((size_t)ROWS * DMAIN);  // doubles as Zh later (no overlap in time)
  ws.CMPh = halloc((size_t)ROWS * DENC);

  _Float16* embedh = halloc((size_t)VOCABSZ * DENC);
  _Float16* wqh    = halloc((size_t)DENC * DENC);
  _Float16* wkh    = halloc((size_t)DENC * DENC);
  _Float16* pupH   = halloc((size_t)DMAIN * DENC);
  _Float16* pdnH   = halloc((size_t)DENC * DMAIN);
  _Float16 *encW[4], *mnW[4], *decW[4];
  const size_t eQ = (size_t)3 * DENC * DENC, eO = (size_t)DENC * DENC,
               eM = (size_t)2 * DENC * DENC;
  const size_t mQ = (size_t)3 * DMAIN * DMAIN, mO = (size_t)DMAIN * DMAIN,
               mM = (size_t)2 * DMAIN * DMAIN;
  encW[0] = halloc(3 * eQ); encW[1] = halloc(3 * eO);
  encW[2] = halloc(3 * eM); encW[3] = halloc(3 * eM);
  mnW[0]  = halloc(6 * mQ); mnW[1]  = halloc(6 * mO);
  mnW[2]  = halloc(6 * mM); mnW[3]  = halloc(6 * mM);
  decW[0] = halloc(3 * eQ); decW[1] = halloc(3 * eO);
  decW[2] = halloc(3 * eM); decW[3] = halloc(3 * eM);

  dim3 blk(256);
  auto cvt = [&](const float* s, _Float16* d, size_t n) {
    HNet_cvt_kernel<<<256, blk, 0, stream>>>(s, d, (int)n);
  };
  cvt(embed, embedh, (size_t)VOCABSZ * DENC);
  cvt(wq, wqh, (size_t)DENC * DENC);
  cvt(wk, wkh, (size_t)DENC * DENC);
  cvt(proj_up, pupH, (size_t)DMAIN * DENC);
  cvt(proj_down, pdnH, (size_t)DENC * DMAIN);
  cvt(enc[2], encW[0], 3 * eQ); cvt(enc[3], encW[1], 3 * eO);
  cvt(enc[6], encW[2], 3 * eM); cvt(enc[7], encW[3], 3 * eM);
  cvt(mn[2],  mnW[0],  6 * mQ); cvt(mn[3],  mnW[1],  6 * mO);
  cvt(mn[6],  mnW[2],  6 * mM); cvt(mn[7],  mnW[3],  6 * mM);
  cvt(dec[2], decW[0], 3 * eQ); cvt(dec[3], decW[1], 3 * eO);
  cvt(dec[6], decW[2], 3 * eM); cvt(dec[7], decW[3], 3 * eM);

  HNet_embed_kernel<<<ROWS, DENC, 0, stream>>>(byte_ids, embed, ws.X);

  for (int i = 0; i < 3; ++i)
    run_block(stream, ws.X, DENC,
              enc[0] + (size_t)i * DENC, enc[1] + (size_t)i * DENC,
              encW[0] + i * eQ, encW[1] + i * eO,
              enc[4] + (size_t)i * DENC, enc[5] + (size_t)i * DENC,
              encW[2] + i * eM, encW[3] + i * eM, nullptr, ws);

  HNet_cvt_kernel<<<256, blk, 0, stream>>>(ws.X, ws.Xh, ROWS * DENC);
  HNet_gemm_kernel<0, 0><<<dim3(DENC / 64, ROWS / 128), blk, 0, stream>>>(
      ws.Xh, wqh, nullptr, ws.XQ, nullptr, ROWS, DENC, DENC);
  HNet_gemm_kernel<0, 0><<<dim3(DENC / 64, ROWS / 128), blk, 0, stream>>>(
      ws.Xh, wkh, nullptr, ws.XK, nullptr, ROWS, DENC, DENC);
  HNet_route_kernel<<<ROWS / 8, blk, 0, stream>>>(ws.XQ, ws.XK, ws.Pp, ws.BH);
  HNet_scan_kernel<<<BATCH, blk, 0, stream>>>(ws.BH, ws.CIDX, ws.MASKF);
  HNet_fillh_kernel<<<256, blk, 0, stream>>>(ws.CMPh, ROWS * DENC);
  HNet_fill_kernel<<<8, blk, 0, stream>>>(ws.PC, ROWS, 0.f);
  HNet_fill_kernel<<<1, 32, 0, stream>>>(ws.RED, 8, 0.f);
  HNet_scatter_kernel<<<ROWS, DENC, 0, stream>>>(ws.X, ws.Pp, ws.BH, ws.CIDX,
                                                 ws.CMPh, ws.PC);
  HNet_reduce_kernel<<<32, blk, 0, stream>>>(ws.BH, ws.Pp, ws.RED);
  HNet_loss_kernel<<<1, 1, 0, stream>>>(ws.RED, (float*)d_out + (size_t)ROWS * VOCABSZ);

  HNet_gemm_kernel<0, 0><<<dim3(DMAIN / 64, ROWS / 128), blk, 0, stream>>>(
      ws.CMPh, pupH, nullptr, ws.Z, nullptr, ROWS, DMAIN, DENC);
  for (int i = 0; i < 6; ++i)
    run_block(stream, ws.Z, DMAIN,
              mn[0] + (size_t)i * DMAIN, mn[1] + (size_t)i * DMAIN,
              mnW[0] + i * mQ, mnW[1] + i * mO,
              mn[4] + (size_t)i * DMAIN, mn[5] + (size_t)i * DMAIN,
              mnW[2] + i * mM, mnW[3] + i * mM, ws.MASKF, ws);
  HNet_cvt_kernel<<<256, blk, 0, stream>>>(ws.Z, ws.Xh, ROWS * DMAIN);  // Zh
  HNet_gemm_kernel<0, 0><<<dim3(DENC / 64, ROWS / 128), blk, 0, stream>>>(
      ws.Xh, pdnH, nullptr, ws.ZD, nullptr, ROWS, DENC, DMAIN);

  HNet_ema_kernel<<<BATCH, DENC, 0, stream>>>(ws.ZD, ws.PC, ws.ZB);
  HNet_addup_kernel<<<ROWS, DENC, 0, stream>>>(ws.X, ws.ZB, ws.CIDX);

  for (int i = 0; i < 3; ++i)
    run_block(stream, ws.X, DENC,
              dec[0] + (size_t)i * DENC, dec[1] + (size_t)i * DENC,
              decW[0] + i * eQ, decW[1] + i * eO,
              dec[4] + (size_t)i * DENC, dec[5] + (size_t)i * DENC,
              decW[2] + i * eM, decW[3] + i * eM, nullptr, ws);

  HNet_ln_kernel<DENC><<<ROWS / 8, blk, 0, stream>>>(ws.X, norm_w, norm_b, ws.T0h);
  HNet_gemm_kernel<0, 0><<<dim3((VOCABSZ + 63) / 64, ROWS / 128), blk, 0, stream>>>(
      ws.T0h, embedh, nullptr, (float*)d_out, nullptr, ROWS, VOCABSZ, DENC);
}
